// MultiHeadAttentionQuantum_65481071404472
// MI455X (gfx1250) — matrix operand order repaired and checked by reading, not compiled, not run on hardware
//
#include <hip/hip_runtime.h>

typedef __attribute__((ext_vector_type(16))) _Float16 v16h;
typedef __attribute__((ext_vector_type(8)))  float    v8f;

#define E_DIM 384
#define T_DIM 512
#define B_DIM 2
#define M_DIM (B_DIM * T_DIM)   // 1024 rows
#define H_DIM 64
#define DK    6

// ---------------------------------------------------------------------------
// Pack 8 consecutive f32 into half of a v16h fragment
// ---------------------------------------------------------------------------
__device__ __forceinline__ void cvt8(const float* __restrict__ p, v16h& f, int base)
{
    float4 x0 = *(const float4*)(p);
    float4 x1 = *(const float4*)(p + 4);
    f[base + 0] = (_Float16)x0.x; f[base + 1] = (_Float16)x0.y;
    f[base + 2] = (_Float16)x0.z; f[base + 3] = (_Float16)x0.w;
    f[base + 4] = (_Float16)x1.x; f[base + 5] = (_Float16)x1.y;
    f[base + 6] = (_Float16)x1.z; f[base + 7] = (_Float16)x1.w;
}

// ---------------------------------------------------------------------------
// WMMA 32x32 tile GEMM core:  Y[m0:m0+32, n0:n0+32] = X @ W^T
// 2 A frags x 2 B frags -> 4 independent accumulators: 4 v_wmma per k-step
// with no D->C dependency among them, hiding the WMMA->WMMA (1 NOP + 4
// coexec) and WMMA->VALU (4 coexec) hazards, at 1 fragment-load per WMMA.
//
// A frag (16-bit A 16x32): lane<16 -> row M=lane&15, K {k0..+7, k0+16..+23};
//                          lane>=16 -> same rows, K shifted by 8.
// B frag (16-bit B 32x16): lane<16 -> col N=lane, K {k0..+7, k0+16..+23};
//                          lane>=16 -> col N=lane-16, K shifted by 8.
// C/D: VGPR j -> M = j + 8*(lane>=16), N = lane&15.
// ---------------------------------------------------------------------------
__device__ __forceinline__ void wmma_tile_32x32(const float* __restrict__ X,
                                                const float* __restrict__ W,
                                                int m0, int n0, int lane,
                                                v8f& acc00, v8f& acc01,
                                                v8f& acc10, v8f& acc11)
{
    const int half = lane >> 4;
    const int r    = lane & 15;

    const float* xrow0 = X + (size_t)(m0 + r) * E_DIM;
    const float* xrow1 = X + (size_t)(m0 + 16 + r) * E_DIM;
    const float* w0row = W + (size_t)(n0 + r) * E_DIM;
    const float* w1row = W + (size_t)(n0 + 16 + r) * E_DIM;

    #pragma unroll 2
    for (int k0 = 0; k0 < E_DIM; k0 += 32) {
        const int ka = k0 + half * 8;
        const int kb = k0 + 16 + half * 8;
        const int kc = k0 + 16 * half;

        v16h a0, a1, b0, b1;
        cvt8(xrow0 + ka, a0, 0);
        cvt8(xrow0 + kb, a0, 8);
        cvt8(xrow1 + ka, a1, 0);
        cvt8(xrow1 + kb, a1, 8);
        cvt8(w0row + ka,     b0, 0);
        cvt8(w0row + kb,     b0, 8);
        cvt8(w1row + ka,     b1, 0);
        cvt8(w1row + kb,     b1, 8);

        acc00 = __builtin_amdgcn_wmma_f32_16x16x32_f16(
            false, a0, false, b0, (short)0, acc00, false, false);
        acc01 = __builtin_amdgcn_wmma_f32_16x16x32_f16(
            false, a0, false, b1, (short)0, acc01, false, false);
        acc10 = __builtin_amdgcn_wmma_f32_16x16x32_f16(
            false, a1, false, b0, (short)0, acc10, false, false);
        acc11 = __builtin_amdgcn_wmma_f32_16x16x32_f16(
            false, a1, false, b1, (short)0, acc11, false, false);
    }
}

__device__ __forceinline__ void store_tile(float* __restrict__ Y, v8f acc,
                                           const float* __restrict__ bias,
                                           int m0, int n0, int lane)
{
    const int half = lane >> 4;
    const int r    = lane & 15;
    const float bn = bias[n0 + r];
    #pragma unroll
    for (int j = 0; j < 8; ++j) {
        int m = m0 + j + 8 * half;
        Y[(size_t)m * E_DIM + n0 + r] = acc[j] + bn;
    }
}

__device__ __forceinline__ void gemm_xwt_body(const float* __restrict__ X,
                                              const float* __restrict__ W,
                                              const float* __restrict__ bias,
                                              float* __restrict__ Y)
{
    const int lane = threadIdx.x & 31;
    const int wave = threadIdx.x >> 5;
    // 2x2 waves cover a 64x64 macro-tile; each wave owns 32x32
    const int m0 = blockIdx.y * 64 + (wave >> 1) * 32;
    const int n0 = blockIdx.x * 64 + (wave & 1) * 32;

    v8f acc00 = {}, acc01 = {}, acc10 = {}, acc11 = {};
    wmma_tile_32x32(X, W, m0, n0, lane, acc00, acc01, acc10, acc11);
    store_tile(Y, acc00, bias, m0,      n0,      lane);
    store_tile(Y, acc01, bias, m0,      n0 + 16, lane);
    store_tile(Y, acc10, bias, m0 + 16, n0,      lane);
    store_tile(Y, acc11, bias, m0 + 16, n0 + 16, lane);
}

// ---------------------------------------------------------------------------
// Stage 1: fused Q/K/V projections. grid = (E/64, M/64, 3), block = 128.
// ---------------------------------------------------------------------------
__global__ __launch_bounds__(128)
void mhaq_proj_gemm(const float* __restrict__ X,
                    const float* __restrict__ Wq, const float* __restrict__ bq,
                    const float* __restrict__ Wk, const float* __restrict__ bk,
                    const float* __restrict__ Wv, const float* __restrict__ bv,
                    float* __restrict__ Q, float* __restrict__ K, float* __restrict__ V)
{
    const int which = blockIdx.z;
    const float* W    = (which == 0) ? Wq : (which == 1) ? Wk : Wv;
    const float* bias = (which == 0) ? bq : (which == 1) ? bk : bv;
    float*       Y    = (which == 0) ? Q  : (which == 1) ? K  : V;
    gemm_xwt_body(X, W, bias, Y);
}

// ---------------------------------------------------------------------------
// Stage 2: closed-form quantum layer, in place over [Q|K|V] (contiguous).
// The circuit before the CNOT ring leaves a product state; CNOTs permute the
// computational basis, so each PauliZ expval is a product of per-wire
//   z_w = cos(a_w)cos(b_w)cos(x_w) - sin(b_w)sin(x_w)
//   out_w = prod_{j<=w} z_j (w>=1),  out_0 = z1*z2*z3*z4*z5
// ---------------------------------------------------------------------------
__global__ __launch_bounds__(256)
void mhaq_qlayer(float* __restrict__ QKV, const float* __restrict__ qp, int ngroups)
{
    int gid = blockIdx.x * blockDim.x + threadIdx.x;
    if (gid >= ngroups) return;
    float* g = QKV + (size_t)gid * DK;

    float z[DK];
    #pragma unroll
    for (int w = 0; w < DK; ++w) {
        float alpha = qp[2 * w];
        float beta  = qp[2 * w + 1];
        float A  = __cosf(alpha) * __cosf(beta);
        float Bc = -__sinf(beta);
        float x  = g[w];
        z[w] = A * __cosf(x) + Bc * __sinf(x);
    }
    float o1 = z[0] * z[1];
    float o2 = o1 * z[2];
    float o3 = o2 * z[3];
    float o4 = o3 * z[4];
    float o5 = o4 * z[5];
    float o0 = z[1] * z[2] * z[3] * z[4] * z[5];
    g[0] = o0; g[1] = o1; g[2] = o2; g[3] = o3; g[4] = o4; g[5] = o5;
}

// ---------------------------------------------------------------------------
// Stage 3: flash-style attention, one workgroup per (b,h). K/V rows of the
// head staged in LDS (24 KB); online softmax per query thread. Writes directly
// into merged (B,T,E) layout.
// ---------------------------------------------------------------------------
__global__ __launch_bounds__(256)
void mhaq_attn(const float* __restrict__ Q, const float* __restrict__ K,
               const float* __restrict__ V, float* __restrict__ Mrg)
{
    __shared__ float ks[T_DIM][DK];
    __shared__ float vs[T_DIM][DK];

    const int bh = blockIdx.x;
    const int b = bh / H_DIM, h = bh % H_DIM;
    const int cbase = h * DK;

    for (int idx = threadIdx.x; idx < T_DIM * DK; idx += blockDim.x) {
        int s = idx / DK, w = idx % DK;
        size_t off = ((size_t)(b * T_DIM + s)) * E_DIM + cbase + w;
        ks[s][w] = K[off];
        vs[s][w] = V[off];
    }
    __syncthreads();

    const float scale = 0.408248290463863f;   // 1/sqrt(6)
    for (int t = threadIdx.x; t < T_DIM; t += blockDim.x) {
        size_t qoff = ((size_t)(b * T_DIM + t)) * E_DIM + cbase;
        float q0 = Q[qoff + 0], q1 = Q[qoff + 1], q2 = Q[qoff + 2];
        float q3 = Q[qoff + 3], q4 = Q[qoff + 4], q5 = Q[qoff + 5];

        float m = -1e30f, l = 0.f;
        float acc[DK] = {0.f, 0.f, 0.f, 0.f, 0.f, 0.f};
        for (int s = 0; s < T_DIM; ++s) {
            float sc = (q0 * ks[s][0] + q1 * ks[s][1] + q2 * ks[s][2] +
                        q3 * ks[s][3] + q4 * ks[s][4] + q5 * ks[s][5]) * scale;
            float nm   = fmaxf(m, sc);
            float corr = __expf(m - nm);
            float p    = __expf(sc - nm);
            l = l * corr + p;
            #pragma unroll
            for (int w = 0; w < DK; ++w) acc[w] = acc[w] * corr + p * vs[s][w];
            m = nm;
        }
        float inv = 1.0f / l;
        #pragma unroll
        for (int w = 0; w < DK; ++w) Mrg[qoff + w] = acc[w] * inv;
    }
}

// ---------------------------------------------------------------------------
// Stage 4: output projection  out = merged @ Wo^T + bo
// ---------------------------------------------------------------------------
__global__ __launch_bounds__(128)
void mhaq_out_gemm(const float* __restrict__ X, const float* __restrict__ W,
                   const float* __restrict__ bias, float* __restrict__ Y)
{
    gemm_xwt_body(X, W, bias, Y);
}

extern "C" void kernel_launch(void* const* d_in, const int* in_sizes, int n_in,
                              void* d_out, int out_size, void* d_ws, size_t ws_size,
                              hipStream_t stream)
{
    const float* x   = (const float*)d_in[0];
    const float* Wq  = (const float*)d_in[1];
    const float* bq  = (const float*)d_in[2];
    const float* Wk  = (const float*)d_in[3];
    const float* bk  = (const float*)d_in[4];
    const float* Wv  = (const float*)d_in[5];
    const float* bv  = (const float*)d_in[6];
    const float* Wo  = (const float*)d_in[7];
    const float* bo  = (const float*)d_in[8];
    const float* qp  = (const float*)d_in[9];

    const size_t mat = (size_t)M_DIM * E_DIM;   // 1024*384 floats
    float* Q   = (float*)d_ws;
    float* K   = Q + mat;
    float* V   = K + mat;
    float* Mrg = V + mat;

    // Stage 1: Q/K/V projections (WMMA f16 in / f32 acc), 32x32 per wave
    dim3 gp(E_DIM / 64, M_DIM / 64, 3);
    mhaq_proj_gemm<<<gp, 128, 0, stream>>>(x, Wq, bq, Wk, bk, Wv, bv, Q, K, V);

    // Stage 2: closed-form quantum transform, in place over contiguous [Q|K|V]
    const int ngroups = 3 * M_DIM * H_DIM;      // 196608
    mhaq_qlayer<<<(ngroups + 255) / 256, 256, 0, stream>>>(Q, qp, ngroups);

    // Stage 3: per-(b,h) fused attention
    mhaq_attn<<<B_DIM * H_DIM, 256, 0, stream>>>(Q, K, V, Mrg);

    // Stage 4: output projection (WMMA)
    dim3 go(E_DIM / 64, M_DIM / 64, 1);
    mhaq_out_gemm<<<go, 128, 0, stream>>>(Mrg, Wo, bo, (float*)d_out);
}
